// DepthSeparableConv2d_conv3_5_37572373905422
// MI455X (gfx1250) — matrix operand order repaired and checked by reading, not compiled, not run on hardware
//
#include <hip/hip_runtime.h>

// ---- types matching gfx1250 WMMA builtin signatures ----
typedef __bf16 bf16x4 __attribute__((ext_vector_type(4)));
typedef __bf16 bf16x8 __attribute__((ext_vector_type(8)));
typedef __bf16 v16bf  __attribute__((ext_vector_type(16)));
typedef float  v8f    __attribute__((ext_vector_type(8)));

#define N_   32
#define CIN  256
#define COUT 512
#define HH   56
#define HW   3136     // 56*56
#define DW_TH 4.0f
#define PW_TH 1e-3f
#define HWP  3140     // LDS row stride (3140 % 8 == 4 breaks M vs M+8 bank aliasing)
#define SMEM2 (16*HWP*4 + 16*4 + 16*4 + 512*4 + 16*4)

// -------------------------------------------------------------------------
// Kernel 1: depthwise 3x3 + bias + BN1 + ReLU + per-(n,c) channel cut.
// One block per (n,c) plane -> whole-plane max needs no atomics.
// Output: bf16, layout [n][c][hw].
// -------------------------------------------------------------------------
__global__ void __launch_bounds__(256) dw_kernel(
    const float* __restrict__ x, const float* __restrict__ dww,
    const float* __restrict__ dwb, const float* __restrict__ g1,
    const float* __restrict__ b1, const float* __restrict__ m1,
    const float* __restrict__ v1, __bf16* __restrict__ y) {
  __shared__ float tile[58 * 58];
  __shared__ float red[256];
  const int bid = blockIdx.x;
  const int n = bid >> 8, c = bid & 255;
  const int tid = threadIdx.x;

  const float* xp = x + (size_t)(n * CIN + c) * HW;
  for (int i = tid; i < 58 * 58; i += 256) {
    int r = i / 58 - 1, col = i % 58 - 1;
    float val = 0.f;
    if (r >= 0 && r < HH && col >= 0 && col < HH) val = xp[r * HH + col];
    tile[i] = val;
  }
  float w[9];
#pragma unroll
  for (int k = 0; k < 9; ++k) w[k] = dww[c * 9 + k];
  const float rs = rsqrtf(v1[c] + 1e-5f);
  const float sc = g1[c] * rs;
  const float sh = b1[c] - m1[c] * sc + dwb[c] * sc;  // bias folded into BN shift
  __syncthreads();

  float vreg[13];
  float vmax = 0.f;
#pragma unroll
  for (int k = 0; k < 13; ++k) {
    int pix = tid + k * 256;
    float val = 0.f;
    if (pix < HW) {
      int r = pix / HH, col = pix % HH;
      const float* t = &tile[r * 58 + col];
      float acc = t[0]   * w[0] + t[1]   * w[1] + t[2]   * w[2]
                + t[58]  * w[3] + t[59]  * w[4] + t[60]  * w[5]
                + t[116] * w[6] + t[117] * w[7] + t[118] * w[8];
      val = fmaxf(acc * sc + sh, 0.f);   // post-ReLU, so max|.| == max
      vmax = fmaxf(vmax, val);
    }
    vreg[k] = val;
  }
  red[tid] = vmax;
  __syncthreads();
  for (int s = 128; s > 0; s >>= 1) {
    if (tid < s) red[tid] = fmaxf(red[tid], red[tid + s]);
    __syncthreads();
  }
  const float keep = (red[0] < DW_TH) ? 0.f : 1.f;

  __bf16* yp = y + (size_t)(n * CIN + c) * HW;
#pragma unroll
  for (int k = 0; k < 13; ++k) {
    int pix = tid + k * 256;
    if (pix < HW) yp[pix] = (__bf16)(vreg[k] * keep);
  }
}

// -------------------------------------------------------------------------
// Kernel 2: pw_w fp32 -> bf16  (512x256, stays hot in L2)
// -------------------------------------------------------------------------
__global__ void wcvt_kernel(const float* __restrict__ pw, __bf16* __restrict__ wb) {
  int i = blockIdx.x * 256 + threadIdx.x;  // 131072 total
  wb[i] = (__bf16)pw[i];
}

// -------------------------------------------------------------------------
// Kernel 3: transpose bf16 [n][c][hw] -> [n][hw][c] so GEMM B-operand lane
// loads (K=c contiguous per hw column) are contiguous 16B loads.
// -------------------------------------------------------------------------
__global__ void __launch_bounds__(256) tr_kernel(const __bf16* __restrict__ y,
                                                 __bf16* __restrict__ yt) {
  __shared__ __bf16 t[32][33];
  const int bid = blockIdx.x;
  const int n = bid / 784;       // 98 hw-tiles * 8 c-tiles
  const int rem = bid % 784;
  const int h0 = (rem / 8) * 32, c0 = (rem % 8) * 32;
  const int row = threadIdx.x >> 3, col4 = (threadIdx.x & 7) * 4;

  const __bf16* src = y + ((size_t)(n * CIN + c0) + row) * HW + h0 + col4;
  bf16x4 in4 = *(const bf16x4*)src;
  t[row][col4 + 0] = in4[0];
  t[row][col4 + 1] = in4[1];
  t[row][col4 + 2] = in4[2];
  t[row][col4 + 3] = in4[3];
  __syncthreads();

  bf16x4 o;
  o[0] = t[col4 + 0][row];
  o[1] = t[col4 + 1][row];
  o[2] = t[col4 + 2][row];
  o[3] = t[col4 + 3][row];
  __bf16* dst = yt + ((size_t)n * HW + h0 + row) * CIN + c0 + col4;
  *(bf16x4*)dst = o;
}

// -------------------------------------------------------------------------
// Kernel 4: pointwise GEMM via v_wmma_f32_16x16x32_bf16 + BN2 + ReLU +
// per-(n,o) channel cut. One workgroup (512 threads = 16 waves, 4 waves per
// SIMD32 for latency hiding) owns one (n, 16-wide o-tile) and the FULL
// 3136-pixel plane, buffered in ~200KB LDS (CDNA5: 320KB/WGP), so the
// channel cut needs no atomics and no extra global pass.
// -------------------------------------------------------------------------
__global__ void __launch_bounds__(512, 1) pw_gemm_kernel(
    const __bf16* __restrict__ yt, const __bf16* __restrict__ wb,
    const float* __restrict__ pwb, const float* __restrict__ g2,
    const float* __restrict__ b2, const float* __restrict__ m2,
    const float* __restrict__ v2, float* __restrict__ z) {
  extern __shared__ char smem[];
  float* zbuf   = (float*)smem;                     // [16][HWP]
  float* coefS  = (float*)(smem + 16 * HWP * 4);    // [16]
  float* coefB  = coefS + 16;                       // [16]
  float* red    = coefB + 16;                       // [512]
  float* rowkeep= red + 512;                        // [16]

  const int bid = blockIdx.x;
  const int n = bid >> 5, o0 = (bid & 31) * 16;
  const int tid = threadIdx.x;

  if (tid < 16) {
    int o = o0 + tid;
    float rs = rsqrtf(v2[o] + 1e-5f);
    float s = g2[o] * rs;
    coefS[tid] = s;
    coefB[tid] = b2[o] - m2[o] * s + pwb[o] * s;  // pw bias folded in
  }
  __syncthreads();

  const int wave = tid >> 5, lane = tid & 31;
  const int lane16 = lane & 15, hi = lane >> 4;

  // Preload all 8 A fragments (ISA A layout: lanes<16 hold K{0..7,16..23},
  // lanes>=16 hold K{8..15,24..31} of row M=lane16).
  v16bf afrag[8];
  {
    const __bf16* arow = wb + (size_t)(o0 + lane16) * CIN;
#pragma unroll
    for (int ks = 0; ks < 8; ++ks) {
      const __bf16* p = arow + ks * 32 + hi * 8;
      bf16x8 lo = *(const bf16x8*)(p);
      bf16x8 hg = *(const bf16x8*)(p + 16);
      afrag[ks] = __builtin_shufflevector(lo, hg, 0, 1, 2, 3, 4, 5, 6, 7, 8,
                                          9, 10, 11, 12, 13, 14, 15);
    }
  }
  float s2l[8], b2l[8];
#pragma unroll
  for (int r = 0; r < 8; ++r) {
    s2l[r] = coefS[r + hi * 8];
    b2l[r] = coefB[r + hi * 8];
  }

  const __bf16* ybase = yt + (size_t)n * HW * CIN;

  // 196 hw-tiles of 16 striped over 16 waves
  for (int jt = wave; jt < 196; jt += 16) {
    const int j0 = jt * 16;
    // B layout: lanes<16 = column j0+lane16 K{0..7,16..23}, lanes>=16 = K{8..15,24..31}
    const __bf16* bcol = ybase + (size_t)(j0 + lane16) * CIN + hi * 8;
    // pull next tile of this wave toward cache while we compute this one
    if (jt + 16 < 196)
      __builtin_prefetch(bcol + (size_t)16 * 16 * CIN, 0, 1);
    v8f acc = {};
#pragma unroll
    for (int ks = 0; ks < 8; ++ks) {
      bf16x8 lo = *(const bf16x8*)(bcol + ks * 32);
      bf16x8 hg = *(const bf16x8*)(bcol + ks * 32 + 16);
      v16bf bfrag = __builtin_shufflevector(lo, hg, 0, 1, 2, 3, 4, 5, 6, 7, 8,
                                            9, 10, 11, 12, 13, 14, 15);
      acc = __builtin_amdgcn_wmma_f32_16x16x32_bf16(
          false, afrag[ks], false, bfrag, (short)0, acc, false, false);
    }
    // epilogue -> LDS (C layout: VGPR r holds M=r (lanes<16) / M=r+8)
    const int col = j0 + lane16;
#pragma unroll
    for (int r = 0; r < 8; ++r) {
      float val = fmaxf(acc[r] * s2l[r] + b2l[r], 0.f);
      zbuf[(r + hi * 8) * HWP + col] = val;
    }
  }
  __syncthreads();

  // per-o-row max over the whole plane (values are post-ReLU, >= 0)
  {
    const int orow = tid >> 5, t32 = tid & 31;
    float m = 0.f;
    for (int j = t32; j < HW; j += 32) m = fmaxf(m, zbuf[orow * HWP + j]);
    red[tid] = m;
    __syncthreads();
    if (t32 == 0) {
      float mm = red[tid];
#pragma unroll
      for (int q = 1; q < 32; ++q) mm = fmaxf(mm, red[tid + q]);
      rowkeep[orow] = (mm < PW_TH) ? 0.f : 1.f;
    }
  }
  __syncthreads();

  float* zp = z + (size_t)(n * COUT + o0) * HW;
  for (int idx = tid; idx < 16 * HW; idx += 512) {
    int ol = idx / HW, j = idx - ol * HW;
    zp[(size_t)ol * HW + j] = zbuf[ol * HWP + j] * rowkeep[ol];
  }
}

// -------------------------------------------------------------------------
extern "C" void kernel_launch(void* const* d_in, const int* in_sizes, int n_in,
                              void* d_out, int out_size, void* d_ws,
                              size_t ws_size, hipStream_t stream) {
  (void)in_sizes; (void)n_in; (void)out_size; (void)ws_size;
  const float* x   = (const float*)d_in[0];
  const float* dww = (const float*)d_in[1];
  const float* dwb = (const float*)d_in[2];
  const float* g1  = (const float*)d_in[3];
  const float* b1  = (const float*)d_in[4];
  const float* m1  = (const float*)d_in[5];
  const float* v1  = (const float*)d_in[6];
  const float* pw  = (const float*)d_in[7];
  const float* pwb = (const float*)d_in[8];
  const float* g2  = (const float*)d_in[9];
  const float* b2  = (const float*)d_in[10];
  const float* m2  = (const float*)d_in[11];
  const float* v2  = (const float*)d_in[12];
  float* z = (float*)d_out;

  // yA (bf16 [n][c][hw], 51.4MB) lives in the front of d_out: it is fully
  // written by dw_kernel, consumed by tr_kernel, then d_out is fully
  // overwritten by pw_gemm_kernel. Stream ordering makes this safe and
  // deterministic. yT + wB live in workspace (~51.6MB).
  __bf16* yA = (__bf16*)d_out;
  __bf16* yT = (__bf16*)d_ws;
  __bf16* wB = (__bf16*)((char*)d_ws + (size_t)N_ * CIN * HW * 2);

  // allow ~200KB dynamic LDS (CDNA5 supports 320KB per workgroup)
  hipFuncSetAttribute((const void*)pw_gemm_kernel,
                      hipFuncAttributeMaxDynamicSharedMemorySize, SMEM2);

  dw_kernel<<<N_ * CIN, 256, 0, stream>>>(x, dww, dwb, g1, b1, m1, v1, yA);
  wcvt_kernel<<<COUT * CIN / 256, 256, 0, stream>>>(pw, wB);
  tr_kernel<<<N_ * 784, 256, 0, stream>>>(yA, yT);
  pw_gemm_kernel<<<N_ * (COUT / 16), 512, SMEM2, stream>>>(yT, wB, pwb, g2, b2,
                                                           m2, v2, z);
}
